// YOSOAttention_63926293233879
// MI455X (gfx1250) — matrix operand order repaired and checked by reading, not compiled, not run on hardware
//
#include <hip/hip_runtime.h>
#include <math.h>

typedef __attribute__((ext_vector_type(16))) _Float16 v16h;
typedef __attribute__((ext_vector_type(8)))  _Float16 v8h;
typedef __attribute__((ext_vector_type(8)))  float    v8f;

#define B_   2
#define H_   12
#define S_   2048
#define D_   64
#define EPSF 1e-6f
#define TQ   128            // Q rows per block
#define NWAVE (TQ / 16)     // 8 waves of 32 lanes

// ---------------------------------------------------------------------------
// w(sim) = (1 - acos(clip(sim))/pi)^h, evaluated to (better than) f16
// accuracy in ~14 VALU ops, branch-free:
//   acos(a) ~= sqrt(1-a) * (C0 + C1 a + C2 a^2 + C3 a^3),  a = |x|
//   (Abramowitz-Stegun 4.4.45, |err| <= 6.7e-5 on [0,1])
//   1 - acos(x)/pi = 0.5 + copysign(0.5 - acos(|x|)/pi, x)
//   pow via single-instruction v_log_f32 / v_exp_f32 (w in (0,1), so log ok)
// ---------------------------------------------------------------------------
__device__ __forceinline__ float yoso_w(float sim, float hfl) {
    const float inv_pi = 0.31830988618379067f;
    float x = __builtin_amdgcn_fmed3f(sim, -1.f + EPSF, 1.f - EPSF);
    float a = __builtin_fabsf(x);
    float p = __builtin_fmaf(a, -0.0187293f, 0.0742610f);
    p = __builtin_fmaf(a, p, -0.2121144f);
    p = __builtin_fmaf(a, p, 1.5707288f);
    float r = __builtin_amdgcn_sqrtf(1.f - a) * p;          // acos(|x|)
    float u = __builtin_fmaf(-r, inv_pi, 0.5f);             // 0.5 - acos(|x|)/pi
    float w = 0.5f + __builtin_copysignf(u, x);             // in (0,1)
    return __builtin_amdgcn_exp2f(hfl * __builtin_amdgcn_logf(w));
}

// ---------------------------------------------------------------------------
// Pass 1: L2-normalize Q,K -> f16 (row-major [bh][s][d]);
//         convert V -> f16 transposed Vt[bh][d][s], pre-multiplied by the
//         key mask (folds the key mask out of the attention inner loop).
// ---------------------------------------------------------------------------
__global__ void yoso_prep(const float* __restrict__ Q, const float* __restrict__ K,
                          const float* __restrict__ V, const int* __restrict__ mask,
                          _Float16* __restrict__ Qh, _Float16* __restrict__ Kh,
                          _Float16* __restrict__ Vt) {
    const int row = blockIdx.x * blockDim.x + threadIdx.x;   // 0 .. B*H*S-1
    if (row >= B_ * H_ * S_) return;
    const int s   = row % S_;
    const int bh  = row / S_;
    const int b   = bh / H_;
    const size_t base = (size_t)row * D_;

    {   // Q normalize
        float ss = 0.f;
        #pragma unroll
        for (int i = 0; i < D_; ++i) { float v = Q[base + i]; ss += v * v; }
        const float inv = rsqrtf(ss + EPSF);
        #pragma unroll
        for (int i = 0; i < D_; ++i) Qh[base + i] = (_Float16)(Q[base + i] * inv);
    }
    {   // K normalize
        float ss = 0.f;
        #pragma unroll
        for (int i = 0; i < D_; ++i) { float v = K[base + i]; ss += v * v; }
        const float inv = rsqrtf(ss + EPSF);
        #pragma unroll
        for (int i = 0; i < D_; ++i) Kh[base + i] = (_Float16)(K[base + i] * inv);
    }
    {   // V transpose-convert with key mask folded in: Vt[bh][d][s]
        const float km = (float)mask[b * S_ + s];
        _Float16* vt = Vt + (size_t)bh * D_ * S_ + s;
        #pragma unroll
        for (int d = 0; d < D_; ++d) vt[(size_t)d * S_] = (_Float16)(V[base + d] * km);
    }
}

// ---------------------------------------------------------------------------
// Pass 2: fused YOSO attention. One wave = 16 Q rows; block = 128 Q rows.
// ---------------------------------------------------------------------------
__global__ __launch_bounds__(NWAVE * 32)
void yoso_attn(const _Float16* __restrict__ Qh, const _Float16* __restrict__ Kh,
               const _Float16* __restrict__ Vt, const int* __restrict__ mask,
               const int* __restrict__ hlen_p, float* __restrict__ out) {
    // per-wave 16x32 f16 w-tile used to re-layout C-layout -> A-layout
    __shared__ _Float16 wlds[NWAVE][16][32];

    const int tid  = threadIdx.x;
    const int lane = tid & 31;
    const int wv   = tid >> 5;
    const int l16  = lane & 15;
    const bool hi  = lane >= 16;

    const int tiles = S_ / TQ;
    const int bh    = blockIdx.x / tiles;
    const int tile  = blockIdx.x % tiles;
    const int b     = bh / H_;
    const int qbase = tile * TQ + wv * 16;
    const float hfl = (float)hlen_p[0];          // dynamic exponent as float

    // --- resident A fragments for Qn (16x32, contiguous k = 16h + s order, not the ISA table order) ---
    // lanes 0-15: M=lane, K in {0..15}; lanes 16-31: M=lane-16, K in {16..31}
    const _Float16* qrow = Qh + ((size_t)bh * S_ + qbase + l16) * D_;
    const int aoff = hi ? 16 : 0;
    v16h aq0, aq1;
    {
        v8h* a0 = (v8h*)&aq0;
        v8h* a1 = (v8h*)&aq1;
        a0[0] = *(const v8h*)(qrow + aoff);          // d 0..7   (or 16..23)
        a0[1] = *(const v8h*)(qrow + 8 + aoff);      // d 8..15  (or 24..31)
        a1[0] = *(const v8h*)(qrow + 32 + aoff);     // d 32..39 (or 48..55)
        a1[1] = *(const v8h*)(qrow + 40 + aoff);     // d 40..47 (or 56..63)
    }

    v8f xacc0 = {}, xacc1 = {}, xacc2 = {}, xacc3 = {};
    v8f* xacc[4] = { &xacc0, &xacc1, &xacc2, &xacc3 };

    const int doff = hi ? 16 : 0;
    // per-lane base pointers (advance by constants inside the loop)
    const _Float16* kbase = Kh + ((size_t)bh * S_ + l16) * D_;
    const _Float16* vbase = Vt + ((size_t)bh * D_ + l16) * S_ + (hi ? 16 : 0);

    for (int kt = 0; kt < S_; kt += 32) {
        // ---- preload all four K B-fragments (2 subtiles x 2 halves) ----
        const _Float16* krow0 = kbase + (size_t)kt * D_;         // keys kt..kt+15
        const _Float16* krow1 = krow0 + 16 * D_;                 // keys kt+16..kt+31
        v16h bk00 = *(const v16h*)(krow0 + doff);
        v16h bk01 = *(const v16h*)(krow0 + 32 + doff);
        v16h bk10 = *(const v16h*)(krow1 + doff);
        v16h bk11 = *(const v16h*)(krow1 + 32 + doff);
        if (kt + 32 < S_) {   // prefetch next K tile (global_prefetch_b8)
            __builtin_prefetch((const void*)(krow0 + 32 * D_), 0, 1);
        }

        // ---- sim = Qn . Kn^T : 16x16 f32 tiles for both key subtiles ----
        v8f c0 = {}, c1 = {};
        c0 = __builtin_amdgcn_wmma_f32_16x16x32_f16(false, aq0, false, bk00,
                                                    (short)0, c0, false, false);
        c0 = __builtin_amdgcn_wmma_f32_16x16x32_f16(false, aq1, false, bk01,
                                                    (short)0, c0, false, false);
        c1 = __builtin_amdgcn_wmma_f32_16x16x32_f16(false, aq0, false, bk10,
                                                    (short)0, c1, false, false);
        c1 = __builtin_amdgcn_wmma_f32_16x16x32_f16(false, aq1, false, bk11,
                                                    (short)0, c1, false, false);

        // ---- cheap branch-free w-transform, store to LDS w-tile ----
        const int mrow = hi ? 8 : 0;
        #pragma unroll
        for (int g = 0; g < 8; ++g) {
            wlds[wv][g + mrow][l16]      = (_Float16)yoso_w(c0[g], hfl);
        }
        #pragma unroll
        for (int g = 0; g < 8; ++g) {
            wlds[wv][g + mrow][16 + l16] = (_Float16)yoso_w(c1[g], hfl);
        }

        // ---- re-load w as A fragment (16x32, K = 32 keys) ----
        v16h wa;
        {
            v8h* w2 = (v8h*)&wa;
            const _Float16* wr = &wlds[wv][l16][0];
            w2[0] = *(const v8h*)(wr + aoff);
            w2[1] = *(const v8h*)(wr + 8 + aoff);
        }

        // ---- preload all four V B-fragments, then 4 WMMAs for d=0..63 ----
        const _Float16* vb = vbase + kt;
        v16h bv0 = *(const v16h*)(vb);
        v16h bv1 = *(const v16h*)(vb + (size_t)16 * S_);
        v16h bv2 = *(const v16h*)(vb + (size_t)32 * S_);
        v16h bv3 = *(const v16h*)(vb + (size_t)48 * S_);
        xacc0 = __builtin_amdgcn_wmma_f32_16x16x32_f16(false, wa, false, bv0,
                                                       (short)0, xacc0, false, false);
        xacc1 = __builtin_amdgcn_wmma_f32_16x16x32_f16(false, wa, false, bv1,
                                                       (short)0, xacc1, false, false);
        xacc2 = __builtin_amdgcn_wmma_f32_16x16x32_f16(false, wa, false, bv2,
                                                       (short)0, xacc2, false, false);
        xacc3 = __builtin_amdgcn_wmma_f32_16x16x32_f16(false, wa, false, bv3,
                                                       (short)0, xacc3, false, false);
    }

    // ---- query mask, row L2-normalize, store f32 output ----
    #pragma unroll
    for (int g = 0; g < 8; ++g) {
        const int q  = qbase + g + (hi ? 8 : 0);
        const float qm = (float)mask[b * S_ + q];
        float ss = 0.f;
        #pragma unroll
        for (int t = 0; t < 4; ++t) {
            float x = (*xacc[t])[g] * qm;
            (*xacc[t])[g] = x;
            ss += x * x;
        }
        // reduce across the 16 lanes holding this row (xor of bits 0..3
        // never crosses the 16-lane half boundary)
        #pragma unroll
        for (int off = 1; off < 16; off <<= 1) ss += __shfl_xor(ss, off, 32);
        const float inv = rsqrtf(ss + EPSF);
        float* orow = out + ((size_t)bh * S_ + q) * D_ + l16;
        #pragma unroll
        for (int t = 0; t < 4; ++t) orow[t * 16] = (*xacc[t])[g] * inv;
    }
}

// ---------------------------------------------------------------------------
extern "C" void kernel_launch(void* const* d_in, const int* in_sizes, int n_in,
                              void* d_out, int out_size, void* d_ws, size_t ws_size,
                              hipStream_t stream) {
    (void)in_sizes; (void)n_in; (void)out_size; (void)ws_size;

    const float* Q    = (const float*)d_in[0];
    const float* K    = (const float*)d_in[1];
    const float* V    = (const float*)d_in[2];
    const int*   mask = (const int*)d_in[3];
    const int*   hlen = (const int*)d_in[4];
    float*       out  = (float*)d_out;

    const size_t N = (size_t)B_ * H_ * S_ * D_;     // 3,145,728 elements
    _Float16* Qh = (_Float16*)d_ws;                 // N halfs
    _Float16* Kh = Qh + N;                          // N halfs
    _Float16* Vt = Kh + N;                          // N halfs (18.9 MB total)

    const int rows = B_ * H_ * S_;
    yoso_prep<<<(rows + 255) / 256, 256, 0, stream>>>(Q, K, V, mask, Qh, Kh, Vt);

    const int grid = B_ * H_ * (S_ / TQ);           // 384 blocks, 8 waves each
    yoso_attn<<<grid, NWAVE * 32, 0, stream>>>(Qh, Kh, Vt, mask, hlen, out);
}
